// LearnableAlignment_58737972740532
// MI455X (gfx1250) — matrix operand order repaired and checked by reading, not compiled, not run on hardware
//
#include <hip/hip_runtime.h>
#include <stdint.h>

// ---------------------------------------------------------------------------
// Types for CDNA5 WMMA (wave32)
// ---------------------------------------------------------------------------
typedef __bf16 bf16;
typedef __attribute__((ext_vector_type(16))) __bf16       v16bf;
typedef __attribute__((ext_vector_type(8)))  float        v8f;
typedef __attribute__((ext_vector_type(4)))  unsigned int u32x4;
typedef __attribute__((ext_vector_type(2)))  unsigned int u32x2;
typedef __attribute__((ext_vector_type(8)))  int          i32x8;
typedef __attribute__((ext_vector_type(4)))  int          i32x4;

union FragAB { v16bf v; u32x4 q[2]; };

__device__ __forceinline__ v8f wmma_bf16(const FragAB& a, const FragAB& b, v8f c) {
  return __builtin_amdgcn_wmma_f32_16x16x32_bf16(false, a.v, false, b.v,
                                                 (short)0, c, false, false);
}

#if defined(__has_builtin)
#if __has_builtin(__builtin_amdgcn_tensor_load_to_lds)
#define HAVE_TDM 1
#endif
#endif
#ifndef HAVE_TDM
#define HAVE_TDM 0
#endif

#if HAVE_TDM
// Tensor Data Mover: 2D tile (rows x rowlen bf16, global row stride) -> LDS.
// D# per CDNA5 ISA ch.8: group0 {count=1, lds_addr, global_addr, type=2},
// group1 {data_size=2B, tensor dims, tile dims, dim0 stride}.
__device__ __forceinline__ void tdm_load_2d(unsigned lds_off, const void* gaddr,
                                            int rows, int rowlen_elems,
                                            int row_stride_elems) {
  unsigned long long ga = (unsigned long long)(uintptr_t)gaddr;
  u32x4 g0 = (u32x4)0u;
  g0[0] = 1u;                                          // count=1 (valid D#)
  g0[1] = lds_off;                                     // lds_addr (bytes)
  g0[2] = (unsigned)(ga & 0xffffffffu);                // global_addr[31:0]
  g0[3] = (unsigned)((ga >> 32) & 0x1ffffffu) | (2u << 30);  // addr[56:32], type=2
  i32x8 g1 = (i32x8)0;
  g1[0] = 1 << 16;                                     // data_size=1 -> 2 bytes
  g1[1] = (row_stride_elems & 0xffff) << 16;           // tensor_dim0[15:0]
  g1[2] = ((row_stride_elems >> 16) & 0xffff) |        // tensor_dim0[31:16]
          ((rows & 0xffff) << 16);                     // tensor_dim1[15:0]
  g1[3] = (rowlen_elems & 0xffff) << 16;               // tile_dim0
  g1[4] = rows & 0xffff;                               // tile_dim1 (tile_dim2=0)
  g1[5] = row_stride_elems;                            // tensor_dim0_stride[31:0]
  i32x4 z4 = (i32x4)0;
#if __clang_major__ >= 23
  i32x8 z8 = (i32x8)0;
  __builtin_amdgcn_tensor_load_to_lds(g0, g1, z4, z4, z8, 0);
#else
  __builtin_amdgcn_tensor_load_to_lds(g0, g1, z4, z4, 0);
#endif
}
#endif

#define CC    256
#define DFF   512
#define HC    56
#define NSEQ  3136       // 56*56
#define HIMG  224
#define PIX   50176      // 224*224
#define KCONV 2304       // 9*256
#define BATCH 2
#define MROWS 6272       // BATCH*NSEQ

// ---------------------------------------------------------------------------
// Prep kernels
// ---------------------------------------------------------------------------
__global__ void cvt_seq_kernel(const float* __restrict__ x, bf16* __restrict__ o) {
  size_t i = (size_t)blockIdx.x * 256 + threadIdx.x;   // over (b,c,p)
  int p = (int)(i % PIX);
  size_t r = i / PIX;
  int c = (int)(r % CC);
  int b = (int)(r / CC);
  o[((size_t)b * PIX + p) * CC + c] = (bf16)x[i];
}

__global__ void prep_convw_kernel(const float* __restrict__ w, bf16* __restrict__ o) {
  int i = blockIdx.x * 256 + threadIdx.x;   // 256*2304
  int co = i / KCONV, k = i % KCONV;
  int tap = k >> 8, ci = k & 255;
  int ky = tap / 3, kx = tap % 3;
  o[i] = (bf16)w[((co * CC + ci) * 3 + ky) * 3 + kx];
}

__global__ void prep_bn_kernel(const float* __restrict__ cb, const float* __restrict__ g,
                               const float* __restrict__ bb, const float* __restrict__ m,
                               const float* __restrict__ v, float* __restrict__ s,
                               float* __restrict__ t) {
  int c = threadIdx.x;
  float inv = g[c] * rsqrtf(v[c] + 1e-5f);
  s[c] = inv;
  t[c] = (cb[c] - m[c]) * inv + bb[c];
}

__global__ void prep_matT_kernel(const float* __restrict__ W, bf16* __restrict__ o,
                                 int K, int N) {
  int i = blockIdx.x * 256 + threadIdx.x;
  if (i >= K * N) return;
  int n = i / K, k = i % K;
  o[i] = (bf16)W[k * N + n];
}

__global__ void prep_flat_kernel(const float* __restrict__ W, bf16* __restrict__ o, int n) {
  int i = blockIdx.x * 256 + threadIdx.x;
  if (i < n) o[i] = (bf16)W[i];
}

// ---------------------------------------------------------------------------
// Stride-4 3x3 conv as implicit GEMM (M=6272, N=256, K=2304) + BN + ReLU
// block = 256 threads (8 waves), tile M=16 x N=128, K steps of 64.
// Double-buffered: TDM fills stage s^1 while WMMAs consume stage s.
// ---------------------------------------------------------------------------
__global__ void conv4_wmma_kernel(const bf16* __restrict__ xseq,   // (B,PIX,C) bf16
                                  const bf16* __restrict__ wB,     // [256][2304]
                                  const float* __restrict__ sc, const float* __restrict__ bi,
                                  bf16* __restrict__ outseq)       // (B,NSEQ,C)
{
  __shared__ bf16 As[2][16 * 64];
  __shared__ bf16 Bs[2][128 * 64];
  const int nblk = blockIdx.x;
  const int m0 = blockIdx.y * 16;
  const int tid = threadIdx.x;
  const int wave = tid >> 5, lane = tid & 31;
  const int g = lane >> 4, ln16 = lane & 15;
  const bool lead = (__builtin_amdgcn_readfirstlane(wave) == 0);
#if HAVE_TDM
  const unsigned bs_off[2] = { (unsigned)(uintptr_t)&Bs[0][0],
                               (unsigned)(uintptr_t)&Bs[1][0] };
#endif

  auto loadB = [&](int st, int k0) {
#if HAVE_TDM
    if (lead) tdm_load_2d(bs_off[st], wB + (size_t)(nblk * 128) * KCONV + k0, 128, 64, KCONV);
#else
    int n = tid >> 1, kh = (tid & 1) * 32;
    const u32x4* src = (const u32x4*)(wB + (size_t)(nblk * 128 + n) * KCONV + k0 + kh);
    u32x4* dst = (u32x4*)(Bs[st] + n * 64 + kh);
    dst[0] = src[0]; dst[1] = src[1]; dst[2] = src[2]; dst[3] = src[3];
#endif
  };
  auto waitB = [&]() {
#if HAVE_TDM
    if (lead) __builtin_amdgcn_s_wait_tensorcnt(0);
#endif
  };
  auto loadA = [&](int st, int k0) {  // im2col gather: 4 consecutive k per thread
    int e = tid * 4;
    int ar = e >> 6, ak = e & 63;
    int m = m0 + ar;
    int b = m / NSEQ, p = m % NSEQ;
    int oy = p / HC, ox = p % HC;
    int k = k0 + ak;
    int tap = k >> 8, ci = k & 255;
    int ky = tap / 3, kx = tap % 3;
    int iy = oy * 4 + ky - 1, ix = ox * 4 + kx - 1;
    u32x2 vv = (u32x2)0u;
    if ((unsigned)iy < HIMG && (unsigned)ix < HIMG)
      vv = *(const u32x2*)(xseq + ((size_t)b * PIX + iy * HIMG + ix) * CC + ci);
    *(u32x2*)(As[st] + ar * 64 + ak) = vv;
  };

  v8f acc;
#pragma unroll
  for (int r = 0; r < 8; ++r) acc[r] = 0.f;

  // prologue: fill stage 0
  loadB(0, 0);
  loadA(0, 0);
  waitB();
  __syncthreads();

  int st = 0;
  for (int k0 = 0; k0 < KCONV; k0 += 64) {
    int k1 = k0 + 64;
    if (k1 < KCONV) {          // prefetch next stage while computing this one
      loadB(st ^ 1, k1);
      loadA(st ^ 1, k1);
    }
#pragma unroll
    for (int s = 0; s < 2; ++s) {
      FragAB a, b;
      const bf16* ap = As[st] + ln16 * 64 + s * 32 + g * 8;
      a.q[0] = *(const u32x4*)ap;
      a.q[1] = *(const u32x4*)(ap + 16);
      const bf16* bp = Bs[st] + (wave * 16 + ln16) * 64 + s * 32 + g * 8;
      b.q[0] = *(const u32x4*)bp;
      b.q[1] = *(const u32x4*)(bp + 16);
      acc = wmma_bf16(a, b, acc);
    }
    if (k1 < KCONV) {
      waitB();
      __syncthreads();
    }
    st ^= 1;
  }

  int n = nblk * 128 + wave * 16 + ln16;
  float s = sc[n], t = bi[n];
#pragma unroll
  for (int r = 0; r < 8; ++r) {
    int m = m0 + r + 8 * g;
    float v = acc[r] * s + t;
    v = v > 0.f ? v : 0.f;
    outseq[(size_t)m * CC + n] = (bf16)v;
  }
}

// ---------------------------------------------------------------------------
// Generic WMMA GEMM: out = epilogue(A[MxK] @ Bt^T + bias)
// A bf16 row-major [M][K]; Bt bf16 [N][K]; tile M=16 x N=128, 8 waves,
// K steps of 64; double-buffered TDM B staging.
// MODE: 0=bf16  1=bf16+relu  2=bf16 V-transposed  3=f32+bf16 residual
//       4=f32+f32 residual  6=f32
// ---------------------------------------------------------------------------
template <int MODE>
__global__ void gemm_wmma_kernel(const bf16* __restrict__ A, const bf16* __restrict__ Bt,
                                 const float* __restrict__ bias, const void* __restrict__ res,
                                 void* __restrict__ out, int K, int N)
{
  __shared__ bf16 As[2][16 * 64];
  __shared__ bf16 Bs[2][128 * 64];
  const int nblk = blockIdx.x;
  const int m0 = blockIdx.y * 16;
  const int tid = threadIdx.x;
  const int wave = tid >> 5, lane = tid & 31;
  const int g = lane >> 4, ln16 = lane & 15;
  const bool lead = (__builtin_amdgcn_readfirstlane(wave) == 0);
#if HAVE_TDM
  const unsigned bs_off[2] = { (unsigned)(uintptr_t)&Bs[0][0],
                               (unsigned)(uintptr_t)&Bs[1][0] };
#endif

  auto loadB = [&](int st, int k0) {
#if HAVE_TDM
    if (lead) tdm_load_2d(bs_off[st], Bt + (size_t)(nblk * 128) * K + k0, 128, 64, K);
#else
    int n = tid >> 1, kh = (tid & 1) * 32;
    const u32x4* src = (const u32x4*)(Bt + (size_t)(nblk * 128 + n) * K + k0 + kh);
    u32x4* dst = (u32x4*)(Bs[st] + n * 64 + kh);
    dst[0] = src[0]; dst[1] = src[1]; dst[2] = src[2]; dst[3] = src[3];
#endif
  };
  auto waitB = [&]() {
#if HAVE_TDM
    if (lead) __builtin_amdgcn_s_wait_tensorcnt(0);
#endif
  };
  auto loadA = [&](int st, int k0) {
    int e = tid * 4;
    int ar = e >> 6, ak = e & 63;
    *(u32x2*)(As[st] + ar * 64 + ak) =
        *(const u32x2*)(A + (size_t)(m0 + ar) * K + k0 + ak);
  };

  v8f acc;
#pragma unroll
  for (int r = 0; r < 8; ++r) acc[r] = 0.f;

  loadB(0, 0);
  loadA(0, 0);
  waitB();
  __syncthreads();

  int st = 0;
  for (int k0 = 0; k0 < K; k0 += 64) {
    int k1 = k0 + 64;
    if (k1 < K) {
      loadB(st ^ 1, k1);
      loadA(st ^ 1, k1);
    }
#pragma unroll
    for (int s = 0; s < 2; ++s) {
      FragAB a, b;
      const bf16* ap = As[st] + ln16 * 64 + s * 32 + g * 8;
      a.q[0] = *(const u32x4*)ap;
      a.q[1] = *(const u32x4*)(ap + 16);
      const bf16* bp = Bs[st] + (wave * 16 + ln16) * 64 + s * 32 + g * 8;
      b.q[0] = *(const u32x4*)bp;
      b.q[1] = *(const u32x4*)(bp + 16);
      acc = wmma_bf16(a, b, acc);
    }
    if (k1 < K) {
      waitB();
      __syncthreads();
    }
    st ^= 1;
  }

  int n = nblk * 128 + wave * 16 + ln16;
  float bsv = bias[n];
#pragma unroll
  for (int r = 0; r < 8; ++r) {
    int m = m0 + r + 8 * g;
    float v = acc[r] + bsv;
    if (MODE == 1) v = fmaxf(v, 0.f);
    if (MODE == 0 || MODE == 1) {
      ((bf16*)out)[(size_t)m * N + n] = (bf16)v;
    } else if (MODE == 2) {  // V transposed: vt[b][n][j]
      int b = m / NSEQ, j = m % NSEQ;
      ((bf16*)out)[((size_t)b * CC + n) * NSEQ + j] = (bf16)v;
    } else if (MODE == 3) {
      v += (float)((const bf16*)res)[(size_t)m * N + n];
      ((float*)out)[(size_t)m * N + n] = v;
    } else if (MODE == 4) {
      v += ((const float*)res)[(size_t)m * N + n];
      ((float*)out)[(size_t)m * N + n] = v;
    } else if (MODE == 6) {
      ((float*)out)[(size_t)m * N + n] = v;
    }
  }
}

// ---------------------------------------------------------------------------
// Flash cross-attention: block = 4 waves, each wave owns 16 query rows.
// ---------------------------------------------------------------------------
__global__ void attn_wmma_kernel(const bf16* __restrict__ Q, const bf16* __restrict__ Km,
                                 const bf16* __restrict__ Vt, bf16* __restrict__ ctx)
{
  __shared__ bf16 slab[4][16 * 32];
  const int tid = threadIdx.x;                 // 0..127
  const int wave = tid >> 5, lane = tid & 31;
  const int g = lane >> 4, ln16 = lane & 15;
  int blk = blockIdx.x;                        // b*8*49 blocks
  int b = blk / (8 * 49);
  int rem = blk % (8 * 49);
  int h = rem / 49;
  int q0 = (rem % 49) * 64 + wave * 16;

  FragAB qa;
  {
    const bf16* qp = Q + ((size_t)(b * NSEQ + q0 + ln16) * CC + h * 32 + g * 8);
    qa.q[0] = *(const u32x4*)qp;
    qa.q[1] = *(const u32x4*)(qp + 16);
  }
  v8f acc0, acc1;
  float mrow[8], lrow[8];
#pragma unroll
  for (int r = 0; r < 8; ++r) { acc0[r] = 0.f; acc1[r] = 0.f; mrow[r] = -1e30f; lrow[r] = 0.f; }

  const float scale = 0.17677669529663688f;    // 1/sqrt(32)
  bf16* sl = slab[wave];

  for (int j0 = 0; j0 < NSEQ; j0 += 32) {
    FragAB kf0, kf1;
    const bf16* kp0 = Km + ((size_t)(b * NSEQ + j0 + ln16) * CC + h * 32 + g * 8);
    kf0.q[0] = *(const u32x4*)kp0;
    kf0.q[1] = *(const u32x4*)(kp0 + 16);
    const bf16* kp1 = kp0 + 16 * CC;
    kf1.q[0] = *(const u32x4*)kp1;
    kf1.q[1] = *(const u32x4*)(kp1 + 16);
    // prefetch next K tile (global_prefetch_b8)
    __builtin_prefetch(kp0 + 32 * CC, 0, 1);

    v8f z;
#pragma unroll
    for (int r = 0; r < 8; ++r) z[r] = 0.f;
    v8f s0 = wmma_bf16(qa, kf0, z);
    v8f s1 = wmma_bf16(qa, kf1, z);

#pragma unroll
    for (int r = 0; r < 8; ++r) {
      float a0 = s0[r] * scale, a1 = s1[r] * scale;
      float mx = fmaxf(a0, a1);
      mx = fmaxf(mx, __shfl_xor(mx, 1));
      mx = fmaxf(mx, __shfl_xor(mx, 2));
      mx = fmaxf(mx, __shfl_xor(mx, 4));
      mx = fmaxf(mx, __shfl_xor(mx, 8));
      float mnew = fmaxf(mrow[r], mx);
      float corr = __expf(mrow[r] - mnew);
      float p0 = __expf(a0 - mnew);
      float p1 = __expf(a1 - mnew);
      float rs = p0 + p1;
      rs += __shfl_xor(rs, 1);
      rs += __shfl_xor(rs, 2);
      rs += __shfl_xor(rs, 4);
      rs += __shfl_xor(rs, 8);
      lrow[r] = lrow[r] * corr + rs;
      mrow[r] = mnew;
      acc0[r] *= corr;
      acc1[r] *= corr;
      int row = r + 8 * g;
      sl[row * 32 + ln16] = (bf16)p0;
      sl[row * 32 + ln16 + 16] = (bf16)p1;
    }

    FragAB pa;
    const bf16* pp = sl + ln16 * 32 + g * 8;
    pa.q[0] = *(const u32x4*)pp;
    pa.q[1] = *(const u32x4*)(pp + 16);

    FragAB vf0, vf1;
    const bf16* vp0 = Vt + ((size_t)(b * CC + h * 32 + ln16) * NSEQ + j0 + g * 8);
    vf0.q[0] = *(const u32x4*)vp0;
    vf0.q[1] = *(const u32x4*)(vp0 + 16);
    const bf16* vp1 = vp0 + (size_t)16 * NSEQ;
    vf1.q[0] = *(const u32x4*)vp1;
    vf1.q[1] = *(const u32x4*)(vp1 + 16);
    __builtin_prefetch(vp0 + 32, 0, 1);

    acc0 = wmma_bf16(pa, vf0, acc0);
    acc1 = wmma_bf16(pa, vf1, acc1);
  }

#pragma unroll
  for (int r = 0; r < 8; ++r) {
    int row = r + 8 * g;
    float inv = 1.f / lrow[r];
    size_t base = (size_t)(b * NSEQ + q0 + row) * CC + h * 32;
    ctx[base + ln16] = (bf16)(acc0[r] * inv);
    ctx[base + 16 + ln16] = (bf16)(acc1[r] * inv);
  }
}

// ---------------------------------------------------------------------------
// LayerNorm over rows of 256: block = 8 waves, one wave per row
// ---------------------------------------------------------------------------
__global__ void ln_rows_kernel(const float* __restrict__ y, const float* __restrict__ gg,
                               const float* __restrict__ bb, bf16* __restrict__ ob,
                               float* __restrict__ of)
{
  int row = blockIdx.x * 8 + (threadIdx.x >> 5);
  int lane = threadIdx.x & 31;
  const float* p = y + (size_t)row * CC;
  float vals[8];
  float sum = 0.f;
#pragma unroll
  for (int i = 0; i < 8; ++i) { vals[i] = p[lane + i * 32]; sum += vals[i]; }
  sum += __shfl_xor(sum, 1);  sum += __shfl_xor(sum, 2);
  sum += __shfl_xor(sum, 4);  sum += __shfl_xor(sum, 8);
  sum += __shfl_xor(sum, 16);
  float mu = sum * (1.f / 256.f);
  float var = 0.f;
#pragma unroll
  for (int i = 0; i < 8; ++i) { float d = vals[i] - mu; var += d * d; }
  var += __shfl_xor(var, 1);  var += __shfl_xor(var, 2);
  var += __shfl_xor(var, 4);  var += __shfl_xor(var, 8);
  var += __shfl_xor(var, 16);
  float inv = rsqrtf(var * (1.f / 256.f) + 1e-5f);
#pragma unroll
  for (int i = 0; i < 8; ++i) {
    int c = lane + i * 32;
    float o = (vals[i] - mu) * inv * gg[c] + bb[c];
    ob[(size_t)row * CC + c] = (bf16)o;
    if (of) of[(size_t)row * CC + c] = o;
  }
}

// ---------------------------------------------------------------------------
// Final: out = bilinear_up(poc) + sp   (op_b folded into poc, sp_b into sp)
// ---------------------------------------------------------------------------
__global__ void final_combine_kernel(const float* __restrict__ poc,  // [b*3136][256]
                                     const bf16* __restrict__ sp,    // [b*50176][256]
                                     float* __restrict__ out)        // (B,C,224,224)
{
  size_t i = (size_t)blockIdx.x * 256 + threadIdx.x;
  int x = (int)(i % HIMG);
  size_t r = i / HIMG;
  int y = (int)(r % HIMG);
  size_t r2 = r / HIMG;
  int co = (int)(r2 % CC);
  int b = (int)(r2 / CC);

  float fy = y * 0.25f - 0.375f;
  float fx = x * 0.25f - 0.375f;
  int y0f = (int)floorf(fy), x0f = (int)floorf(fx);
  float wy = fy - (float)y0f, wx = fx - (float)x0f;
  int y0 = min(max(y0f, 0), HC - 1), y1 = min(max(y0f + 1, 0), HC - 1);
  int x0 = min(max(x0f, 0), HC - 1), x1 = min(max(x0f + 1, 0), HC - 1);

  auto P = [&](int yy, int xx) {
    return poc[((size_t)b * NSEQ + yy * HC + xx) * CC + co];
  };
  float up = (1.f - wy) * ((1.f - wx) * P(y0, x0) + wx * P(y0, x1)) +
             wy * ((1.f - wx) * P(y1, x0) + wx * P(y1, x1));
  float spv = (float)sp[((size_t)b * PIX + y * HIMG + x) * CC + co];
  out[i] = up + spv;
}

// ---------------------------------------------------------------------------
// Host launcher
// ---------------------------------------------------------------------------
extern "C" void kernel_launch(void* const* d_in, const int* in_sizes, int n_in,
                              void* d_out, int out_size, void* d_ws, size_t ws_size,
                              hipStream_t stream) {
  (void)in_sizes; (void)n_in; (void)out_size; (void)ws_size;
  const float* lidar = (const float*)d_in[0];
  const float* image = (const float*)d_in[1];
  const float* cl_w = (const float*)d_in[2];
  const float* cl_b = (const float*)d_in[3];
  const float* bnl_g = (const float*)d_in[4];
  const float* bnl_b = (const float*)d_in[5];
  const float* bnl_m = (const float*)d_in[6];
  const float* bnl_v = (const float*)d_in[7];
  const float* ci_w = (const float*)d_in[8];
  const float* ci_b = (const float*)d_in[9];
  const float* bni_g = (const float*)d_in[10];
  const float* bni_b = (const float*)d_in[11];
  const float* bni_m = (const float*)d_in[12];
  const float* bni_v = (const float*)d_in[13];
  const float* qw = (const float*)d_in[14];
  const float* qb = (const float*)d_in[15];
  const float* kw = (const float*)d_in[16];
  const float* kb = (const float*)d_in[17];
  const float* vw = (const float*)d_in[18];
  const float* vb = (const float*)d_in[19];
  const float* ow = (const float*)d_in[20];
  const float* ob = (const float*)d_in[21];
  const float* ln1_g = (const float*)d_in[22];
  const float* ln1_b = (const float*)d_in[23];
  const float* w1 = (const float*)d_in[24];
  const float* b1 = (const float*)d_in[25];
  const float* w2 = (const float*)d_in[26];
  const float* b2 = (const float*)d_in[27];
  const float* ln2_g = (const float*)d_in[28];
  const float* ln2_b = (const float*)d_in[29];
  const float* op_w = (const float*)d_in[30];
  const float* op_b = (const float*)d_in[31];
  const float* sp_w = (const float*)d_in[32];
  const float* sp_b = (const float*)d_in[33];

  char* wp = (char*)d_ws;
  auto alloc = [&](size_t bytes) -> void* {
    void* p = (void*)wp;
    wp += (bytes + 255) & ~(size_t)255;
    return p;
  };

  bf16* lidar_seq = (bf16*)alloc((size_t)BATCH * PIX * CC * 2);
  bf16* image_seq = (bf16*)alloc((size_t)BATCH * PIX * CC * 2);
  bf16* wBl = (bf16*)alloc((size_t)CC * KCONV * 2);
  bf16* wBi = (bf16*)alloc((size_t)CC * KCONV * 2);
  float* sl = (float*)alloc(CC * 4);
  float* tl = (float*)alloc(CC * 4);
  float* si = (float*)alloc(CC * 4);
  float* ti = (float*)alloc(CC * 4);
  bf16* qwT = (bf16*)alloc((size_t)CC * CC * 2);
  bf16* kwT = (bf16*)alloc((size_t)CC * CC * 2);
  bf16* vwT = (bf16*)alloc((size_t)CC * CC * 2);
  bf16* owT = (bf16*)alloc((size_t)CC * CC * 2);
  bf16* w1T = (bf16*)alloc((size_t)CC * DFF * 2);
  bf16* w2T = (bf16*)alloc((size_t)DFF * CC * 2);
  bf16* opwB = (bf16*)alloc((size_t)CC * CC * 2);
  bf16* spwB = (bf16*)alloc((size_t)CC * CC * 2);
  bf16* lseq = (bf16*)alloc((size_t)MROWS * CC * 2);
  bf16* iseq = (bf16*)alloc((size_t)MROWS * CC * 2);
  bf16* Qm = (bf16*)alloc((size_t)MROWS * CC * 2);
  bf16* Km = (bf16*)alloc((size_t)MROWS * CC * 2);
  bf16* vtV = (bf16*)alloc((size_t)BATCH * CC * NSEQ * 2);
  bf16* ctxm = (bf16*)alloc((size_t)MROWS * CC * 2);
  float* y1 = (float*)alloc((size_t)MROWS * CC * 4);
  bf16* xb = (bf16*)alloc((size_t)MROWS * CC * 2);
  float* xf = (float*)alloc((size_t)MROWS * CC * 4);
  bf16* tffn = (bf16*)alloc((size_t)MROWS * DFF * 2);
  float* y2 = (float*)alloc((size_t)MROWS * CC * 4);
  bf16* ocm = (bf16*)alloc((size_t)MROWS * CC * 2);
  float* poc = (float*)alloc((size_t)MROWS * CC * 4);
  bf16* spv = (bf16*)alloc((size_t)BATCH * PIX * CC * 2);

  // --- prep ---
  cvt_seq_kernel<<<100352, 256, 0, stream>>>(lidar, lidar_seq);
  cvt_seq_kernel<<<100352, 256, 0, stream>>>(image, image_seq);
  prep_convw_kernel<<<2304, 256, 0, stream>>>(cl_w, wBl);
  prep_convw_kernel<<<2304, 256, 0, stream>>>(ci_w, wBi);
  prep_bn_kernel<<<1, 256, 0, stream>>>(cl_b, bnl_g, bnl_b, bnl_m, bnl_v, sl, tl);
  prep_bn_kernel<<<1, 256, 0, stream>>>(ci_b, bni_g, bni_b, bni_m, bni_v, si, ti);
  prep_matT_kernel<<<256, 256, 0, stream>>>(qw, qwT, CC, CC);
  prep_matT_kernel<<<256, 256, 0, stream>>>(kw, kwT, CC, CC);
  prep_matT_kernel<<<256, 256, 0, stream>>>(vw, vwT, CC, CC);
  prep_matT_kernel<<<256, 256, 0, stream>>>(ow, owT, CC, CC);
  prep_matT_kernel<<<512, 256, 0, stream>>>(w1, w1T, CC, DFF);
  prep_matT_kernel<<<512, 256, 0, stream>>>(w2, w2T, DFF, CC);
  prep_flat_kernel<<<256, 256, 0, stream>>>(op_w, opwB, CC * CC);
  prep_flat_kernel<<<256, 256, 0, stream>>>(sp_w, spwB, CC * CC);

  // --- stride-4 conv + BN + ReLU (implicit GEMM, WMMA + TDM pipeline) ---
  conv4_wmma_kernel<<<dim3(2, 392), 256, 0, stream>>>(lidar_seq, wBl, sl, tl, lseq);
  conv4_wmma_kernel<<<dim3(2, 392), 256, 0, stream>>>(image_seq, wBi, si, ti, iseq);

  // --- QKV projections ---
  gemm_wmma_kernel<0><<<dim3(2, 392), 256, 0, stream>>>(lseq, qwT, qb, nullptr, Qm, CC, CC);
  gemm_wmma_kernel<0><<<dim3(2, 392), 256, 0, stream>>>(iseq, kwT, kb, nullptr, Km, CC, CC);
  gemm_wmma_kernel<2><<<dim3(2, 392), 256, 0, stream>>>(iseq, vwT, vb, nullptr, vtV, CC, CC);

  // --- flash cross-attention ---
  attn_wmma_kernel<<<784, 128, 0, stream>>>(Qm, Km, vtV, ctxm);

  // --- out-proj + residual, LN1 ---
  gemm_wmma_kernel<3><<<dim3(2, 392), 256, 0, stream>>>(ctxm, owT, ob, lseq, y1, CC, CC);
  ln_rows_kernel<<<784, 256, 0, stream>>>(y1, ln1_g, ln1_b, xb, xf);

  // --- FFN + residual, LN2 ---
  gemm_wmma_kernel<1><<<dim3(4, 392), 256, 0, stream>>>(xb, w1T, b1, nullptr, tffn, CC, DFF);
  gemm_wmma_kernel<4><<<dim3(2, 392), 256, 0, stream>>>(tffn, w2T, b2, xf, y2, DFF, CC);
  ln_rows_kernel<<<784, 256, 0, stream>>>(y2, ln2_g, ln2_b, ocm, nullptr);

  // --- 1x1 convs (WMMA GEMMs); upsample after conv (they commute) ---
  gemm_wmma_kernel<6><<<dim3(2, 392), 256, 0, stream>>>(ocm, opwB, op_b, nullptr, poc, CC, CC);
  gemm_wmma_kernel<0><<<dim3(2, 6272), 256, 0, stream>>>(lidar_seq, spwB, sp_b, nullptr, spv, CC, CC);

  // --- bilinear upsample + combine ---
  final_combine_kernel<<<100352, 256, 0, stream>>>(poc, spv, (float*)d_out);
}
